// TelephoneAttentionND_41936060678698
// MI455X (gfx1250) — compile-verified, matrix operand order checked by reading, not run on hardware
//
#include <hip/hip_runtime.h>
#include <hip/hip_bf16.h>

// ---------------------------------------------------------------------------
// TelephoneAttentionND for MI455X (gfx1250, wave32, WMMA).
//   B=2, L=2048, C=1024, H=16, D=64, K=64, S=33. All fp32 in/out.
//   - Heavy GEMMs: v_wmma_f32_16x16x32_bf16 with bf16 hi/lo split (3 WMMAs
//     per k-step per tile) -> ~fp32 accuracy at bf16 matrix-core rate.
//     Each wave owns a 16x32 tile (A fragment reused, 6 WMMAs/k-step).
//   - gather_conv stages each wave's 64-float kernel row into LDS with
//     global_load_async_to_lds_b64 (ASYNCcnt) before the data-dependent
//     interpolation gathers.
// ---------------------------------------------------------------------------

typedef __attribute__((ext_vector_type(16))) __bf16 v16bf;
typedef __attribute__((ext_vector_type(8)))  float  v8f;

#define LSEQ   2048
#define CDIM   1024
#define NHEAD  16
#define HDIM   64
#define KSZ    64
#define NSAMP  33

__device__ __forceinline__ float silu(float y) { return y / (1.0f + __expf(-y)); }
__device__ __forceinline__ float sigmoidf(float y) { return 1.0f / (1.0f + __expf(-y)); }

struct Frag2 { v16bf hi, lo; };

// Load a 16-element bf16 A/B fragment (hi + residual lo) for wmma 16x16x32.
// ISA layout (16-bit operand, wave32): lane<16 -> K in {base+0..7, base+16..23}
// with base=0; lane>=16 -> base=8.  p must point at (row, kc + khalf).
// Hardware converts: v_cvt_pk_bf16_f32 + v_pk_add_f32(neg) for the residual.
__device__ __forceinline__ Frag2 load_frag_split(const float* __restrict__ p) {
    v16bf H, L;
#pragma unroll
    for (int i = 0; i < 16; ++i) {
        float a = p[(i < 8) ? i : (i + 8)];     // i>=8 -> +16 + (i-8)
        __bf16 hb = (__bf16)a;
        float lof = a - (float)hb;
        H[i] = hb;
        L[i] = (__bf16)lof;
    }
    Frag2 r; r.hi = H; r.lo = L; return r;
}

// C = act(A(MxK) * W(NxK)^T + bias), fp32 io, bf16x3 WMMA core.
// block = 256 (8 waves); each wave owns a 16x32 tile; grid = (M/128, N/32).
__global__ void __launch_bounds__(256)
gemm_bf16x3(const float* __restrict__ A, const float* __restrict__ W,
            const float* __restrict__ bias, float* __restrict__ C,
            int M, int N, int Kdim, int act) {
    const int lane  = threadIdx.x & 31;
    const int wave  = threadIdx.x >> 5;
    const int m0    = (blockIdx.x * 8 + wave) * 16;
    const int n0    = blockIdx.y * 32;
    if (m0 >= M || n0 >= N) return;

    const int laneM = lane & 15;
    const int khalf = (lane >> 4) * 8;

    const float* arow  = A + (size_t)(m0 + laneM) * Kdim + khalf;
    const float* brow0 = W + (size_t)(n0 + laneM) * Kdim + khalf;
    const float* brow1 = brow0 + (size_t)16 * Kdim;

    v8f acc0 = {0.f, 0.f, 0.f, 0.f, 0.f, 0.f, 0.f, 0.f};
    v8f acc1 = {0.f, 0.f, 0.f, 0.f, 0.f, 0.f, 0.f, 0.f};

    for (int kc = 0; kc < Kdim; kc += 32) {
        if (kc + 32 < Kdim) {                    // hint next tiles into cache
            __builtin_prefetch(arow  + kc + 32, 0, 3);
            __builtin_prefetch(brow0 + kc + 32, 0, 3);
            __builtin_prefetch(brow1 + kc + 32, 0, 3);
        }
        Frag2 a  = load_frag_split(arow  + kc);
        Frag2 b0 = load_frag_split(brow0 + kc);
        Frag2 b1 = load_frag_split(brow1 + kc);
        // C += Ahi*Bhi + Ahi*Blo + Alo*Bhi  (~fp32 accuracy)
        acc0 = __builtin_amdgcn_wmma_f32_16x16x32_bf16(false, a.hi, false, b0.hi,
                                                       (short)0, acc0, false, false);
        acc0 = __builtin_amdgcn_wmma_f32_16x16x32_bf16(false, a.hi, false, b0.lo,
                                                       (short)0, acc0, false, false);
        acc0 = __builtin_amdgcn_wmma_f32_16x16x32_bf16(false, a.lo, false, b0.hi,
                                                       (short)0, acc0, false, false);
        acc1 = __builtin_amdgcn_wmma_f32_16x16x32_bf16(false, a.hi, false, b1.hi,
                                                       (short)0, acc1, false, false);
        acc1 = __builtin_amdgcn_wmma_f32_16x16x32_bf16(false, a.hi, false, b1.lo,
                                                       (short)0, acc1, false, false);
        acc1 = __builtin_amdgcn_wmma_f32_16x16x32_bf16(false, a.lo, false, b1.hi,
                                                       (short)0, acc1, false, false);
    }

    // C/D layout: VGPR r -> (M = r + (lane<16 ? 0 : 8), N = lane&15)
    const int col0    = n0 + laneM;
    const int col1    = col0 + 16;
    const int rowbase = m0 + ((lane >> 4) * 8);
    const float bv0   = bias ? bias[col0] : 0.0f;
    const float bv1   = bias ? bias[col1] : 0.0f;
#pragma unroll
    for (int r = 0; r < 8; ++r) {
        float v0 = acc0[r] + bv0;
        float v1 = acc1[r] + bv1;
        if (act == 1) { v0 = silu(v0); v1 = silu(v1); }
        C[(size_t)(rowbase + r) * N + col0] = v0;
        C[(size_t)(rowbase + r) * N + col1] = v1;
    }
}

// One wave per token: 32-lane wave projection (+RMSNorm over 32, SiLU,
// sigmoid/tanh) and the scalar exponent projection.
__global__ void __launch_bounds__(256)
wave_exp_proj(const float* __restrict__ x,
              const float* __restrict__ w_wave, const float* __restrict__ b_wave,
              const float* __restrict__ g_wave,
              const float* __restrict__ w_exp,  const float* __restrict__ b_exp,
              const float* __restrict__ g_exp,
              float* __restrict__ freqB, float* __restrict__ phaseB,
              float* __restrict__ expB, int T) {
    const int lane = threadIdx.x & 31;
    const int wave = threadIdx.x >> 5;
    const int t = blockIdx.x * 8 + wave;
    if (t >= T) return;

    const float* xr = x + (size_t)t * CDIM;
    const float* wr = w_wave + (size_t)lane * CDIM;
    float acc = 0.f, acce = 0.f;
    for (int k = 0; k < CDIM; ++k) acc += xr[k] * wr[k];
    for (int k = lane; k < CDIM; k += 32) acce += xr[k] * w_exp[k];

    float v  = acc + b_wave[lane];
    float ss = v * v;
#pragma unroll
    for (int m = 16; m; m >>= 1) ss += __shfl_xor(ss, m, 32);
    float y = g_wave[lane] * v * rsqrtf(ss * (1.0f / 32.0f) + 1e-6f);
    y = silu(y);
    if (lane < 16) freqB[t * NHEAD + lane] = 15.0f * sigmoidf(y) + 1.0f;
    else           phaseB[t * NHEAD + (lane - 16)] = tanhf(y) * 16.0f;

#pragma unroll
    for (int m = 16; m; m >>= 1) acce += __shfl_xor(acce, m, 32);
    if (lane == 0) {
        float e = acce + b_exp[0];
        float r = g_exp[0] * e * rsqrtf(e * e + 1e-6f);
        expB[t] = 3.5f * sigmoidf(r) + 0.5f;
    }
}

// Row-wise (bias add +) RMSNorm + SiLU over the 1024-wide kern projection.
// This is the ONLY place b_kern is applied (GEMM step runs with bias=null).
__global__ void __launch_bounds__(256)
rms_silu_kern(const float* __restrict__ P, const float* __restrict__ b,
              const float* __restrict__ g, float* __restrict__ out) {
    const int t = blockIdx.x;
    const float* pr = P + (size_t)t * CDIM;
    __shared__ float red[8];
    float v[4]; float ss = 0.f;
#pragma unroll
    for (int i = 0; i < 4; ++i) {
        int j = threadIdx.x + i * 256;
        v[i] = pr[j] + b[j];
        ss += v[i] * v[i];
    }
#pragma unroll
    for (int m = 16; m; m >>= 1) ss += __shfl_xor(ss, m, 32);
    if ((threadIdx.x & 31) == 0) red[threadIdx.x >> 5] = ss;
    __syncthreads();
    float tot = 0.f;
#pragma unroll
    for (int w = 0; w < 8; ++w) tot += red[w];
    const float rn = rsqrtf(tot * (1.0f / (float)CDIM) + 1e-6f);
#pragma unroll
    for (int i = 0; i < 4; ++i) {
        int j = threadIdx.x + i * 256;
        float y = g[j] * v[i] * rn;
        out[(size_t)t * CDIM + j] = silu(y);
    }
}

// One wave per (token, head).
//   Stage 0: async-copy the wave's 64-float kernel row into LDS
//            (global_load_async_to_lds_b64, 2 floats/lane, ASYNCcnt).
//   Phase 1: lanes compute the 33 per-sample coefficients
//            (LDS kernel interp * power * valid) + floor index + frac -> LDS.
//   Phase 2: each lane accumulates channels {2*lane, 2*lane+1} with float2
//            gathers of the floor/ceil rows from x.
__global__ void __launch_bounds__(256)
gather_conv(const float* __restrict__ x, const float* __restrict__ kmax,
            const float* __restrict__ freqB, const float* __restrict__ phaseB,
            const float* __restrict__ expB, float* __restrict__ hidden, int T) {
    __shared__ float s_kr[8][KSZ];
    __shared__ float s_kc[8][NSAMP];
    __shared__ float s_fr[8][NSAMP];
    __shared__ int   s_fl[8][NSAMP];

    const int lane = threadIdx.x & 31;
    const int wave = threadIdx.x >> 5;
    const int gw = blockIdx.x * 8 + wave;          // (t, h) pair
    const int t = gw >> 4;
    const int h = gw & 15;
    if (t >= T) return;
    const int l = t & (LSEQ - 1);
    const int bidx = t >> 11;                      // L = 2048

    const float freq = freqB[t * NHEAD + h];
    const float phs  = phaseB[t * NHEAD + h];
    const float ex   = expB[t];
    const float* kr  = kmax + ((size_t)t * NHEAD + h) * KSZ;

    // --- Stage 0: coalesced async copy of the kernel row to LDS -----------
    {
        unsigned lds_off = (unsigned)(uintptr_t)(&s_kr[wave][2 * lane]);
        const float* gsrc = kr + 2 * lane;
        asm volatile("global_load_async_to_lds_b64 %0, %1, off"
                     :: "v"(lds_off), "v"(gsrc) : "memory");
        asm volatile("s_wait_asynccnt 0x0" ::: "memory");
    }

#pragma unroll
    for (int rep = 0; rep < 2; ++rep) {
        int s = lane + rep * 32;
        if (s < NSAMP) {
            float off = (float)(s - 16);
            float rel = off * freq;
            float pos = (float)l + rel + phs;
            float valid = (pos >= 0.0f && pos < (float)LSEQ) ? 1.0f : 0.0f;
            float pc = fminf(fmaxf(pos, 0.0f), (float)LSEQ - 1.001f);
            int fl = (int)floorf(pc);
            fl = min(max(fl, 0), LSEQ - 1);
            float frac = pc - (float)fl;
            float ar = fabsf(rel);
            float power = __powf(1.0f + ar * (1.0f / (float)LSEQ), -ex);
            float idx_f = fminf(ar * (1.0f / 256.0f), 1.0f) * (float)(KSZ - 1);
            int i0 = min(max((int)idx_f, 0), KSZ - 2);
            float wc = idx_f - (float)i0;
            float k0 = s_kr[wave][i0], k1 = s_kr[wave][i0 + 1];
            s_kc[wave][s] = (k0 + wc * (k1 - k0)) * power * valid;
            s_fl[wave][s] = fl;
            s_fr[wave][s] = frac;
        }
    }
    __syncthreads();

    const float* xb = x + (size_t)bidx * LSEQ * CDIM + h * HDIM + 2 * lane;
    float ax = 0.f, ay = 0.f;
    for (int s = 0; s < NSAMP; ++s) {
        float kc = s_kc[wave][s];                 // wave-uniform scalar
        if (kc != 0.0f) {
            int   fl = s_fl[wave][s];
            float fr = s_fr[wave][s];
            int   cl = min(fl + 1, LSEQ - 1);
            const float2 vf = *(const float2*)(xb + (size_t)fl * CDIM);
            const float2 vc = *(const float2*)(xb + (size_t)cl * CDIM);
            ax += kc * (vf.x + fr * (vc.x - vf.x));
            ay += kc * (vf.y + fr * (vc.y - vf.y));
        }
    }
    float2 r; r.x = ax; r.y = ay;
    *(float2*)(hidden + (size_t)t * CDIM + h * HDIM + 2 * lane) = r;
}

extern "C" void kernel_launch(void* const* d_in, const int* in_sizes, int n_in,
                              void* d_out, int out_size, void* d_ws, size_t ws_size,
                              hipStream_t stream) {
    const float* x      = (const float*)d_in[0];
    const float* w_wave = (const float*)d_in[1];
    const float* b_wave = (const float*)d_in[2];
    const float* g_wave = (const float*)d_in[3];
    const float* w_kern = (const float*)d_in[4];
    const float* b_kern = (const float*)d_in[5];
    const float* g_kern = (const float*)d_in[6];
    const float* w_exp  = (const float*)d_in[7];
    const float* b_exp  = (const float*)d_in[8];
    const float* g_exp  = (const float*)d_in[9];
    const float* w_out  = (const float*)d_in[10];
    float* out = (float*)d_out;

    const int T = in_sizes[0] / CDIM;              // B*L = 4096 tokens
    const size_t TC = (size_t)T * CDIM;

    float* ws     = (float*)d_ws;
    float* P      = ws;                            // proj-kern result, reused as hidden
    float* kmax   = ws + TC;                       // kernel_max
    float* freqB  = kmax + TC;
    float* phaseB = freqB + (size_t)T * NHEAD;
    float* expB   = phaseB + (size_t)T * NHEAD;

    dim3 blk(256);
    dim3 gg((T + 127) / 128, CDIM / 32);

    // 1. P = x @ w_kern^T                   (WMMA bf16x3; bias applied in step 3)
    gemm_bf16x3<<<gg, blk, 0, stream>>>(x, w_kern, nullptr, P, T, CDIM, CDIM, 0);
    // 2. freq/phase/exponent
    wave_exp_proj<<<dim3((T + 7) / 8), blk, 0, stream>>>(
        x, w_wave, b_wave, g_wave, w_exp, b_exp, g_exp, freqB, phaseB, expB, T);
    // 3. kernel_max = silu(rmsnorm(P + b_kern, g_kern))
    rms_silu_kern<<<dim3(T), blk, 0, stream>>>(P, b_kern, g_kern, kmax);
    // 4. hidden = sum_s kern * val          (overwrites P)
    gather_conv<<<dim3((T * NHEAD + 7) / 8), blk, 0, stream>>>(
        x, kmax, freqB, phaseB, expB, P, T);
    // 5. out = silu(hidden @ w_out^T)       (WMMA bf16x3 + silu)
    gemm_bf16x3<<<gg, blk, 0, stream>>>(P, w_out, nullptr, out, T, CDIM, CDIM, 1);
}
